// NodeModel_19404662243986
// MI455X (gfx1250) — compile-verified, matrix operand order checked by reading, not run on hardware
//
#include <hip/hip_runtime.h>

// ---------------------------------------------------------------------------
// GraphNet NodeModel for gfx1250 (MI455X): WMMA f16->f32 tiles.
//   edge MLP  : [E,128] @ W1a[128,64] -> selu -> @ W1b[64,64] -> scatter-mean
//   node MLP  : [N,192] @ W2a[192,64] -> selu -> @ W2b[64,64] -> out[N,64]
// Fragment loads are 2 x ds_load_b128 each (contiguous-K layout in LDS).
// ---------------------------------------------------------------------------

typedef __attribute__((ext_vector_type(16))) _Float16 v16h;
typedef __attribute__((ext_vector_type(8)))  float    v8f;

#define WMMA16(A, B, C) \
    __builtin_amdgcn_wmma_f32_16x16x32_f16(false, (A), false, (B), (short)0, (C), false, false)

// Load a 16x32 f16 fragment from an LDS tile stored row-major (strideH halves
// per row). Per CDNA5 ISA 7.12.2 (16-bit A-matrix 16x32):
//   lanes 0-15 : VGPR0-3 hold K=0..7 in pairs, VGPR4-7 hold K=16..23
//   lanes 16-31: VGPR0-3 hold K=8..15,         VGPR4-7 hold K=24..31
// Both 4-VGPR groups are 8 consecutive halves -> one ds_load_b128 each.
// B fragments use the same pattern with rowIdx = N (weights stored N-major).
__device__ __forceinline__ v16h ld_frag(const _Float16* base, int strideH,
                                        int kb, int rowIdx, int khi) {
    union { v16h v; uint4 q[2]; } f;
    const _Float16* p = base + rowIdx * strideH + kb + khi * 8;  // 16B aligned
    f.q[0] = *(const uint4*)(p);        // K = kb + khi*8 + 0..7
    f.q[1] = *(const uint4*)(p + 16);   // K = kb + 16 + khi*8 + 0..7
    return f.v;
}

__device__ __forceinline__ float selu_f(float v) {
    const float sc = 1.0507009873554805f;
    const float al = 1.6732632423543772f;
    return v > 0.f ? sc * v : sc * al * (__expf(v) - 1.f);
}

// ---------------------------------------------------------------------------
__global__ void zero_kernel(float* __restrict__ p, long n) {
    long i = (long)blockIdx.x * blockDim.x + threadIdx.x;
    if (i < n) p[i] = 0.f;
}

// ---------------------------------------------------------------------------
// Edge kernel: 8 waves/block, 16 edges per wave.
__global__ void __launch_bounds__(256)
edge_kernel(const float* __restrict__ x, const int* __restrict__ row,
            const int* __restrict__ col, const float* __restrict__ eattr,
            const float* __restrict__ W1a, const float* __restrict__ b1a,
            const float* __restrict__ W1b, const float* __restrict__ b1b,
            float* __restrict__ summed, float* __restrict__ counts, int nE) {
    __shared__ _Float16 sW1a[64 * 128];     // [n][k], N-major
    __shared__ _Float16 sW1b[64 * 64];      // [n][k]
    __shared__ float    sb1a[64], sb1b[64];
    __shared__ _Float16 sStage[8][16 * 128];
    __shared__ int      sCol[8][16];

    const int tid = threadIdx.x;
    // Stage weights (fp32 -> f16, transpose to N-major).
    for (int i = tid; i < 128 * 64; i += 256) {
        int k = i >> 6, n = i & 63;
        sW1a[n * 128 + k] = (_Float16)W1a[i];
    }
    for (int i = tid; i < 64 * 64; i += 256) {
        int k = i >> 6, n = i & 63;
        sW1b[n * 64 + k] = (_Float16)W1b[i];
    }
    if (tid < 64) { sb1a[tid] = b1a[tid]; sb1b[tid] = b1b[tid]; }
    __syncthreads();

    const int wave = tid >> 5, lane = tid & 31;
    const int n_   = lane & 15, khi = lane >> 4;
    const int ebase = (blockIdx.x * 8 + wave) * 16;
    if (ebase >= nE) return;

    _Float16* stage = &sStage[wave][0];
    if (lane < 16) {
        int e = ebase + lane;
        sCol[wave][lane] = (e < nE) ? col[e] : 0;
    }

    // Stage gathered inputs [16][128] as f16: lanes 0-15 pull x[row[e]],
    // lanes 16-31 pull edge_attr[e] (float4 each -> full 128 features).
#pragma unroll 4
    for (int m = 0; m < 16; ++m) {
        int e = ebase + m; if (e >= nE) e = nE - 1;
        const float* src = (lane < 16)
            ? (x + (size_t)row[e] * 64 + lane * 4)
            : (eattr + (size_t)e * 64 + (lane - 16) * 4);
        float4 v = *(const float4*)src;
        union { _Float16 h[4]; uint2 u; } pk;
        pk.h[0] = (_Float16)v.x; pk.h[1] = (_Float16)v.y;
        pk.h[2] = (_Float16)v.z; pk.h[3] = (_Float16)v.w;
        *(uint2*)(stage + m * 128 + lane * 4) = pk.u;
    }
    asm volatile("" ::: "memory");  // keep DS writes before DS reads (in-order per wave)

    // Layer 1: [16,128] x [128,64]
    v16h a[4];
#pragma unroll
    for (int kb = 0; kb < 4; ++kb) a[kb] = ld_frag(stage, 128, kb * 32, n_, khi);
    v8f acc[4]; v8f zero = {};
#pragma unroll
    for (int nb = 0; nb < 4; ++nb) acc[nb] = zero;
#pragma unroll
    for (int kb = 0; kb < 4; ++kb)
#pragma unroll
        for (int nb = 0; nb < 4; ++nb) {
            v16h b = ld_frag(sW1a, 128, kb * 32, nb * 16 + n_, khi);
            acc[nb] = WMMA16(a[kb], b, acc[nb]);
        }

    // selu(acc + b1a) -> restage [16][64] f16 (A-frags already consumed).
    asm volatile("" ::: "memory");
#pragma unroll
    for (int nb = 0; nb < 4; ++nb) {
        int n = nb * 16 + n_;
        float bias = sb1a[n];
#pragma unroll
        for (int r = 0; r < 8; ++r) {
            int m = r + khi * 8;
            stage[m * 64 + n] = (_Float16)selu_f(acc[nb][r] + bias);
        }
    }
    asm volatile("" ::: "memory");

    // Layer 2: [16,64] x [64,64]
    v16h a2[2];
    a2[0] = ld_frag(stage, 64, 0, n_, khi);
    a2[1] = ld_frag(stage, 64, 32, n_, khi);
    v8f acc2[4];
#pragma unroll
    for (int nb = 0; nb < 4; ++nb) acc2[nb] = zero;
#pragma unroll
    for (int kb = 0; kb < 2; ++kb)
#pragma unroll
        for (int nb = 0; nb < 4; ++nb) {
            v16h b = ld_frag(sW1b, 64, kb * 32, nb * 16 + n_, khi);
            acc2[nb] = WMMA16(a2[kb], b, acc2[nb]);
        }

    // Scatter-add into summed[col] (+ counts).
#pragma unroll
    for (int nb = 0; nb < 4; ++nb) {
        int n = nb * 16 + n_;
        float bias = sb1b[n];
#pragma unroll
        for (int r = 0; r < 8; ++r) {
            int m = r + khi * 8;
            if (ebase + m < nE) {
                int dst = sCol[wave][m];
                atomicAdd(&summed[(size_t)dst * 64 + n], acc2[nb][r] + bias);
            }
        }
    }
    if (lane < 16 && ebase + lane < nE)
        atomicAdd(&counts[sCol[wave][lane]], 1.0f);
}

// ---------------------------------------------------------------------------
// Node kernel: 4 waves/block, 16 nodes per wave.
__global__ void __launch_bounds__(128)
node_kernel(const float* __restrict__ x, const float* __restrict__ u,
            const int* __restrict__ batch, const float* __restrict__ W2a,
            const float* __restrict__ b2a, const float* __restrict__ W2b,
            const float* __restrict__ b2b, const float* __restrict__ summed,
            const float* __restrict__ counts, float* __restrict__ out, int nN) {
    __shared__ _Float16 sW2a[64 * 192];     // [n][k]
    __shared__ _Float16 sW2b[64 * 64];
    __shared__ float    sb2a[64], sb2b[64];
    __shared__ _Float16 sStage[4][16 * 192];

    const int tid = threadIdx.x;
    for (int i = tid; i < 192 * 64; i += 128) {
        int k = i >> 6, n = i & 63;
        sW2a[n * 192 + k] = (_Float16)W2a[i];
    }
    for (int i = tid; i < 64 * 64; i += 128) {
        int k = i >> 6, n = i & 63;
        sW2b[n * 64 + k] = (_Float16)W2b[i];
    }
    if (tid < 64) { sb2a[tid] = b2a[tid]; sb2b[tid] = b2b[tid]; }
    __syncthreads();

    const int wave = tid >> 5, lane = tid & 31;
    const int n_   = lane & 15, khi = lane >> 4;
    const int nbase = (blockIdx.x * 4 + wave) * 16;
    if (nbase >= nN) return;

    _Float16* stage = &sStage[wave][0];

    // Build h = [x | summed/max(cnt,1) | u[batch]] as f16 [16][192].
#pragma unroll 2
    for (int m = 0; m < 16; ++m) {
        int node = nbase + m; if (node >= nN) node = nN - 1;
        float invc = 1.f / fmaxf(counts[node], 1.f);
        int g = batch[node];
#pragma unroll
        for (int jj = 0; jj < 6; ++jj) {
            int j = jj * 32 + lane;
            float v;
            if (j < 64)       v = x[(size_t)node * 64 + j];
            else if (j < 128) v = summed[(size_t)node * 64 + (j - 64)] * invc;
            else              v = u[(size_t)g * 64 + (j - 128)];
            stage[m * 192 + j] = (_Float16)v;
        }
    }
    asm volatile("" ::: "memory");

    // Layer 1: [16,192] x [192,64]
    v16h a[6];
#pragma unroll
    for (int kb = 0; kb < 6; ++kb) a[kb] = ld_frag(stage, 192, kb * 32, n_, khi);
    v8f acc[4]; v8f zero = {};
#pragma unroll
    for (int nb = 0; nb < 4; ++nb) acc[nb] = zero;
#pragma unroll
    for (int kb = 0; kb < 6; ++kb)
#pragma unroll
        for (int nb = 0; nb < 4; ++nb) {
            v16h b = ld_frag(sW2a, 192, kb * 32, nb * 16 + n_, khi);
            acc[nb] = WMMA16(a[kb], b, acc[nb]);
        }

    asm volatile("" ::: "memory");
#pragma unroll
    for (int nb = 0; nb < 4; ++nb) {
        int n = nb * 16 + n_;
        float bias = sb2a[n];
#pragma unroll
        for (int r = 0; r < 8; ++r) {
            int m = r + khi * 8;
            stage[m * 64 + n] = (_Float16)selu_f(acc[nb][r] + bias);
        }
    }
    asm volatile("" ::: "memory");

    // Layer 2: [16,64] x [64,64]
    v16h a2[2];
    a2[0] = ld_frag(stage, 64, 0, n_, khi);
    a2[1] = ld_frag(stage, 64, 32, n_, khi);
    v8f acc2[4];
#pragma unroll
    for (int nb = 0; nb < 4; ++nb) acc2[nb] = zero;
#pragma unroll
    for (int kb = 0; kb < 2; ++kb)
#pragma unroll
        for (int nb = 0; nb < 4; ++nb) {
            v16h b = ld_frag(sW2b, 64, kb * 32, nb * 16 + n_, khi);
            acc2[nb] = WMMA16(a2[kb], b, acc2[nb]);
        }

#pragma unroll
    for (int nb = 0; nb < 4; ++nb) {
        int n = nb * 16 + n_;
        float bias = sb2b[n];
#pragma unroll
        for (int r = 0; r < 8; ++r) {
            int m = r + khi * 8;
            int node = nbase + m;
            if (node < nN) out[(size_t)node * 64 + n] = acc2[nb][r] + bias;
        }
    }
}

// ---------------------------------------------------------------------------
extern "C" void kernel_launch(void* const* d_in, const int* in_sizes, int n_in,
                              void* d_out, int out_size, void* d_ws, size_t ws_size,
                              hipStream_t stream) {
    const float* x     = (const float*)d_in[0];
    const int*   ei    = (const int*)  d_in[1];
    const float* eattr = (const float*)d_in[2];
    const float* u     = (const float*)d_in[3];
    const int*   batch = (const int*)  d_in[4];
    const float* W1a   = (const float*)d_in[5];
    const float* b1a   = (const float*)d_in[6];
    const float* W1b   = (const float*)d_in[7];
    const float* b1b   = (const float*)d_in[8];
    const float* W2a   = (const float*)d_in[9];
    const float* b2a   = (const float*)d_in[10];
    const float* W2b   = (const float*)d_in[11];
    const float* b2b   = (const float*)d_in[12];

    const int nN = in_sizes[0] / 64;          // 100000
    const int nE = in_sizes[2] / 64;          // 1600000
    const int* row = ei;
    const int* col = ei + nE;

    float* summed = (float*)d_ws;             // [nN, 64]
    float* counts = summed + (size_t)nN * 64; // [nN]
    float* out    = (float*)d_out;

    long nzero = (long)nN * 64 + nN;
    zero_kernel<<<(int)((nzero + 255) / 256), 256, 0, stream>>>(summed, nzero);

    int edge_blocks = (nE + 127) / 128;       // 128 edges per block (8 waves x 16)
    edge_kernel<<<edge_blocks, 256, 0, stream>>>(x, row, col, eattr,
                                                 W1a, b1a, W1b, b1b,
                                                 summed, counts, nE);

    int node_blocks = (nN + 63) / 64;         // 64 nodes per block (4 waves x 16)
    node_kernel<<<node_blocks, 128, 0, stream>>>(x, u, batch, W2a, b2a, W2b, b2b,
                                                 summed, counts, out, nN);
}
